// HeteroDEMGNN_46626164965861
// MI455X (gfx1250) — matrix operand order repaired and checked by reading, not compiled, not run on hardware
//
#include <hip/hip_runtime.h>
#include <stddef.h>

typedef __attribute__((ext_vector_type(16))) _Float16 v16h;
typedef __attribute__((ext_vector_type(8)))  _Float16 v8h;
typedef __attribute__((ext_vector_type(4)))  _Float16 v4h;
typedef __attribute__((ext_vector_type(8)))  float    v8f;
typedef __attribute__((ext_vector_type(4)))  float    v4f;

enum { MODE_H = 0, MODE_Y = 1, MODE_SCAT = 2, MODE_DEC = 3 };

// ---------------------------------------------------------------------------
// GEMM: Y[M,NOUT] = A[M,KPAD](f16 packed) @ W[K,NOUT](f32) + bias
// MODE_H:    write Y (H buffer, NOUT=64) + BN stats (LDS -> global atomics)
// MODE_Y:    write Y (NOUT=64)
// MODE_SCAT: write Y + atomicAdd scatter rows into scat[dst[m]*64+col]
// MODE_DEC:  Y[m*3+col] += result (NOUT=3)
// 128-thread block = 4 waves x 4 stripes of 16 rows = 256 rows, B staged once
// in LDS in WMMA fragment layout (hoisted to registers by the compiler for
// small KPAD). Bias held in registers. EXEC all-ones through every WMMA.
// M must be a multiple of 16 (holds: N=100000, E=1000000).
// ---------------------------------------------------------------------------
template<int KPAD, int MODE, int NOUT>
__global__ void __launch_bounds__(128)
gemm_kernel(const _Float16* __restrict__ A, const float* __restrict__ W,
            const float* __restrict__ bias, int K, int M,
            float* __restrict__ Y, float* __restrict__ stats,
            const int* __restrict__ dst, float* __restrict__ scat) {
  constexpr int NCH = KPAD / 32;
  __shared__ __align__(32) _Float16 sB[64 * KPAD];
  __shared__ float s_sum[64];
  __shared__ float s_sq[64];
  const int tid = threadIdx.x;
  if constexpr (MODE == MODE_H) {
    if (tid < 64) { s_sum[tid] = 0.0f; s_sq[tid] = 0.0f; }
  }

  // Stage B: fragment(nt,ch) = 32 lanes x 16 halves; lane L = column nt*16+(L&15),
  // element s -> K = ch*32 + (L>=16 ? 8 : 0) + (s<8 ? s : 16+s-8).
  for (int i = tid; i < 64 * KPAD; i += 128) {
    int n = i & 63;
    int kk = i >> 6;
    float w = (kk < K && n < NOUT) ? W[kk * NOUT + n] : 0.0f;
    int frag = (n >> 4) * NCH + (kk >> 5);
    int ln = (((kk >> 3) & 1) << 4) | (n & 15);
    sB[(frag * 32 + ln) * 16 + ((kk & 7) | (((kk >> 4) & 1) << 3))] = (_Float16)w;
  }
  __syncthreads();

  const int wave = tid >> 5;
  const int lane = tid & 31;
  const int n = lane & 15;
  const int rsel = (lane >> 4) << 3;             // 0/8: C-row offset == A K-offset
  const int wbase = (blockIdx.x * 4 + wave) * 64;
  const _Float16* bp = sB + lane * 16;

  // Hoist bias into registers (loop-invariant per lane).
  float bb[4];
  if constexpr (MODE == MODE_DEC) {
    bb[0] = (n < NOUT) ? bias[n] : 0.0f;
    bb[1] = bb[2] = bb[3] = 0.0f;
  } else {
#pragma unroll
    for (int nt = 0; nt < 4; ++nt) bb[nt] = bias[nt * 16 + n];
  }

#pragma unroll 1
  for (int s = 0; s < 4; ++s) {
    int m0 = wbase + s * 16;
    int m0c = (m0 > M - 16) ? (M - 16) : m0;     // clamp tail stripes
    const _Float16* arow = A + (size_t)(m0c + n) * KPAD + rsel;

    v8f acc[4];
#pragma unroll
    for (int q = 0; q < 4; ++q)
#pragma unroll
      for (int r = 0; r < 8; ++r) acc[q][r] = 0.0f;

#pragma unroll
    for (int ch = 0; ch < NCH; ++ch) {
      v8h a0 = *(const v8h*)(arow + ch * 32);
      v8h a1 = *(const v8h*)(arow + ch * 32 + 16);
      v16h a = __builtin_shufflevector(a0, a1, 0, 1, 2, 3, 4, 5, 6, 7,
                                       8, 9, 10, 11, 12, 13, 14, 15);
      v16h b0 = *(const v16h*)(bp + (0 * NCH + ch) * 512);
      v16h b1 = *(const v16h*)(bp + (1 * NCH + ch) * 512);
      v16h b2 = *(const v16h*)(bp + (2 * NCH + ch) * 512);
      v16h b3 = *(const v16h*)(bp + (3 * NCH + ch) * 512);
      acc[0] = __builtin_amdgcn_wmma_f32_16x16x32_f16(false, a, false, b0, (short)0, acc[0], false, false);
      acc[1] = __builtin_amdgcn_wmma_f32_16x16x32_f16(false, a, false, b1, (short)0, acc[1], false, false);
      acc[2] = __builtin_amdgcn_wmma_f32_16x16x32_f16(false, a, false, b2, (short)0, acc[2], false, false);
      acc[3] = __builtin_amdgcn_wmma_f32_16x16x32_f16(false, a, false, b3, (short)0, acc[3], false, false);
    }

    // Epilogue: divergence allowed (WMMAs done). primary => all 16 rows valid.
    bool primary = (m0 == m0c);
    int mbase = m0c + rsel;

    if constexpr (MODE == MODE_DEC) {
      if (primary && n < NOUT) {
        float* p = Y + (size_t)mbase * NOUT + n;
#pragma unroll
        for (int r = 0; r < 8; ++r) {
          *p = *p + acc[0][r] + bb[0];
          p += NOUT;
        }
      }
    } else {
      if (primary) {
        int drow[8];
        if constexpr (MODE == MODE_SCAT) {
#pragma unroll
          for (int r = 0; r < 8; ++r) drow[r] = dst[mbase + r];
        }
        float* yp = Y + ((size_t)mbase << 6) + n;
#pragma unroll
        for (int nt = 0; nt < 4; ++nt) {
          int col = nt * 16 + n;
          float psum = 0.0f, psq = 0.0f;
          float* p = yp + nt * 16;
#pragma unroll
          for (int r = 0; r < 8; ++r) {
            float y = acc[nt][r] + bb[nt];
            p[(size_t)r << 6] = y;
            if constexpr (MODE == MODE_SCAT)
              atomicAdd(&scat[((size_t)drow[r] << 6) + col], y);
            if constexpr (MODE == MODE_H) { psum += y; psq += y * y; }
          }
          if constexpr (MODE == MODE_H) {
            atomicAdd(&s_sum[col], psum);      // ds_add_f32
            atomicAdd(&s_sq[col], psq);
          }
        }
      }
    }
  }

  if constexpr (MODE == MODE_H) {
    __syncthreads();
    if (tid < 64) {
      atomicAdd(&stats[tid], s_sum[tid]);      // global_atomic_add_f32
      atomicAdd(&stats[64 + tid], s_sq[tid]);
    }
  }
}

template<int KPAD, int MODE, int NOUT>
static inline void launch_gemm(hipStream_t st, const _Float16* A, const float* W,
                               const float* b, int K, int M, float* Y,
                               float* stats, const int* dst, float* scat) {
  gemm_kernel<KPAD, MODE, NOUT><<<(M + 255) / 256, 128, 0, st>>>(
      A, W, b, K, M, Y, stats, dst, scat);
}

__global__ void finalize_stats_kernel(const float* __restrict__ stats,
                                      float* __restrict__ musig, float invM) {
  int i = threadIdx.x;                         // 64 threads
  float mu = stats[i] * invM;
  float var = stats[64 + i] * invM - mu * mu;
  musig[i] = mu;
  musig[64 + i] = rsqrtf(var + 1e-5f);
}

// ---------------------------------------------------------------------------
// Pack kernels: 4 columns per thread (16B vector in, 8B half4 out)
// ---------------------------------------------------------------------------
__global__ void pack_pad_kernel(const float* __restrict__ in,
                                _Float16* __restrict__ out,
                                int K, int KPg, size_t total4) {
  size_t i = (size_t)blockIdx.x * blockDim.x + threadIdx.x;
  if (i >= total4) return;
  int g = (int)(i & (((size_t)1 << KPg) - 1));
  size_t m = i >> KPg;
  v4h o;
#pragma unroll
  for (int e = 0; e < 4; ++e) {
    int c = g * 4 + e;
    o[e] = (_Float16)((c < K) ? in[m * K + c] : 0.0f);
  }
  *(v4h*)(out + i * 4) = o;
}

__global__ void pack_cast_kernel(const float* __restrict__ in,
                                 _Float16* __restrict__ out, size_t total4) {
  size_t i = (size_t)blockIdx.x * blockDim.x + threadIdx.x;
  if (i >= total4) return;
  v4f h = *(const v4f*)(in + i * 4);
  v4h o;
#pragma unroll
  for (int e = 0; e < 4; ++e) o[e] = (_Float16)h[e];
  *(v4h*)(out + i * 4) = o;
}

// out[M,64] f16 = relu((H - mu) * rsig)
__global__ void pack_bn_kernel(const float* __restrict__ H,
                               const float* __restrict__ musig,
                               _Float16* __restrict__ out, size_t total4) {
  size_t i = (size_t)blockIdx.x * blockDim.x + threadIdx.x;
  if (i >= total4) return;
  int g = (int)(i & 15);
  v4f h = *(const v4f*)(H + i * 4);
  v4h o;
#pragma unroll
  for (int e = 0; e < 4; ++e) {
    int c = g * 4 + e;
    float v = (h[e] - musig[c]) * musig[64 + c];
    o[e] = (_Float16)fmaxf(v, 0.0f);
  }
  *(v4h*)(out + i * 4) = o;
}

// out[N,128] f16 = [node_emb | aggr]
__global__ void pack_upd_kernel(const float* __restrict__ ne,
                                const float* __restrict__ ag,
                                _Float16* __restrict__ out, size_t total4) {
  size_t i = (size_t)blockIdx.x * blockDim.x + threadIdx.x;
  if (i >= total4) return;
  int g = (int)(i & 31);
  size_t m = i >> 5;
  const float* src = (g < 16) ? (ne + m * 64 + g * 4) : (ag + m * 64 + (g - 16) * 4);
  v4f h = *(const v4f*)src;
  v4h o;
#pragma unroll
  for (int e = 0; e < 4; ++e) o[e] = (_Float16)h[e];
  *(v4h*)(out + i * 4) = o;
}

// out[E,160] f16 = [node_emb[src](64) | ea(4) | ea(4) | eemb(64) | 0(24)]
__global__ void pack_msg_kernel(const float* __restrict__ ne,
                                const int* __restrict__ src,
                                const float* __restrict__ ea,
                                const float* __restrict__ ee,
                                _Float16* __restrict__ out, size_t total4) {
  size_t i = (size_t)blockIdx.x * blockDim.x + threadIdx.x;
  if (i >= total4) return;
  int g = (int)(i % 40);
  size_t e = i / 40;
  v4f h;
  if (g < 16)      h = *(const v4f*)(ne + (size_t)src[e] * 64 + g * 4);
  else if (g < 18) h = *(const v4f*)(ea + e * 4);
  else if (g < 34) h = *(const v4f*)(ee + e * 64 + (size_t)(g - 18) * 4);
  else             h = (v4f){0.0f, 0.0f, 0.0f, 0.0f};
  v4h o;
#pragma unroll
  for (int q = 0; q < 4; ++q) o[q] = (_Float16)h[q];
  *(v4h*)(out + i * 4) = o;
}

// ---------------------------------------------------------------------------
// Small helpers
// ---------------------------------------------------------------------------
__global__ void zero_kernel(float* __restrict__ p, size_t n) {
  size_t i = (size_t)blockIdx.x * blockDim.x + threadIdx.x;
  if (i < n) p[i] = 0.0f;
}

__global__ void count_kernel(const int* __restrict__ dst, int E,
                             int* __restrict__ cnt) {
  int e = blockIdx.x * blockDim.x + threadIdx.x;
  if (e < E) atomicAdd(&cnt[dst[e]], 1);
}

__global__ void aggr_kernel(const float* __restrict__ sum_c, const int* __restrict__ cnt_c,
                            const float* __restrict__ sum_n, const int* __restrict__ cnt_n,
                            float* __restrict__ aggr, int N) {
  int i = blockIdx.x * blockDim.x + threadIdx.x;
  if (i < N * 64) {
    int m = i >> 6;
    float cc = fmaxf((float)cnt_c[m], 1.0f);
    float cn = fmaxf((float)cnt_n[m], 1.0f);
    aggr[i] = 0.5f * (sum_c[i] / cc + sum_n[i] / cn);
  }
}

// ---------------------------------------------------------------------------
// Host orchestration
// ---------------------------------------------------------------------------
extern "C" void kernel_launch(void* const* d_in, const int* in_sizes, int n_in,
                              void* d_out, int out_size, void* d_ws, size_t ws_size,
                              hipStream_t stream) {
  const int N = in_sizes[0] / 3;               // 100000  (multiple of 16)
  const int E = in_sizes[1] / 4;               // 1000000 (multiple of 16)
  const int S = 2;
  (void)n_in; (void)out_size; (void)ws_size;

  const float* x     = (const float*)d_in[0];
  const float* ea_c  = (const float*)d_in[1];
  const float* ea_n  = (const float*)d_in[2];
  const float* nW1   = (const float*)d_in[3];
  const float* nb1   = (const float*)d_in[4];
  const float* nW2   = (const float*)d_in[5];
  const float* nb2   = (const float*)d_in[6];
  const float* eW1   = (const float*)d_in[7];
  const float* eb1   = (const float*)d_in[8];
  const float* eW2   = (const float*)d_in[9];
  const float* eb2   = (const float*)d_in[10];
  const float* msgW1 = (const float*)d_in[11];
  const float* msgb1 = (const float*)d_in[12];
  const float* msgW2 = (const float*)d_in[13];
  const float* msgb2 = (const float*)d_in[14];
  const float* edgW1 = (const float*)d_in[15];
  const float* edgb1 = (const float*)d_in[16];
  const float* edgW2 = (const float*)d_in[17];
  const float* edgb2 = (const float*)d_in[18];
  const float* updW1 = (const float*)d_in[19];
  const float* updb1 = (const float*)d_in[20];
  const float* updW2 = (const float*)d_in[21];
  const float* updb2 = (const float*)d_in[22];
  const float* decW1 = (const float*)d_in[23];
  const float* decb1 = (const float*)d_in[24];
  const float* decW2 = (const float*)d_in[25];
  const float* decb2 = (const float*)d_in[26];
  const int*   ei_c  = (const int*)d_in[27];   // [2,E]: row0=src, row1=dst
  const int*   ei_n  = (const int*)d_in[28];
  float* out = (float*)d_out;

  char* ws = (char*)d_ws;
  auto carve = [&](size_t bytes) -> void* {
    void* p = (void*)ws;
    ws += (bytes + 255) & ~(size_t)255;
    return p;
  };
  float*     node_emb = (float*)carve((size_t)N * 64 * 4);
  float*     H_node   = (float*)carve((size_t)N * 64 * 4);
  float*     aggr     = (float*)carve((size_t)N * 64 * 4);
  float*     sum_c    = (float*)carve((size_t)N * 64 * 4);
  float*     sum_n    = (float*)carve((size_t)N * 64 * 4);
  float*     eemb_c   = (float*)carve((size_t)E * 64 * 4);
  float*     eemb_n   = (float*)carve((size_t)E * 64 * 4);
  float*     msg      = (float*)carve((size_t)E * 64 * 4);
  float*     H_edge   = (float*)carve((size_t)E * 64 * 4);
  _Float16*  P_n      = (_Float16*)carve((size_t)N * 128 * 2);
  _Float16*  P_e160   = (_Float16*)carve((size_t)E * 160 * 2);
  _Float16*  P_e64    = (_Float16*)carve((size_t)E * 64 * 2);
  int*       cnt_c    = (int*)carve((size_t)N * 4);
  int*       cnt_n    = (int*)carve((size_t)N * 4);
  float*     stats    = (float*)carve(128 * 4);
  float*     musig    = (float*)carve(128 * 4);

  auto Z = [&](void* p, size_t nelem) {
    zero_kernel<<<(unsigned)((nelem + 255) / 256), 256, 0, stream>>>((float*)p, nelem);
  };
  auto PAD = [&](const float* in, _Float16* o, int K, int KPg, size_t M_) {
    size_t total4 = M_ << KPg;
    pack_pad_kernel<<<(unsigned)((total4 + 255) / 256), 256, 0, stream>>>(in, o, K, KPg, total4);
  };
  auto CAST = [&](const float* in, _Float16* o, size_t elems) {
    size_t total4 = elems / 4;
    pack_cast_kernel<<<(unsigned)((total4 + 255) / 256), 256, 0, stream>>>(in, o, total4);
  };
  auto BN = [&](const float* H, _Float16* o, size_t M_) {
    size_t total4 = M_ * 16;
    pack_bn_kernel<<<(unsigned)((total4 + 255) / 256), 256, 0, stream>>>(H, musig, o, total4);
  };
  auto STAT0 = [&]() { Z(stats, 128); };
  auto FIN = [&](int M_) {
    finalize_stats_kernel<<<1, 64, 0, stream>>>(stats, musig, 1.0f / (float)M_);
  };

  // --- init ----------------------------------------------------------------
  Z(out, (size_t)N * 3);
  Z(cnt_c, (size_t)N);
  Z(cnt_n, (size_t)N);
  count_kernel<<<(E + 255) / 256, 256, 0, stream>>>(ei_c + E, E, cnt_c);
  count_kernel<<<(E + 255) / 256, 256, 0, stream>>>(ei_n + E, E, cnt_n);

  // --- node embedding MLP: x[N,3] -> node_emb[N,64] ------------------------
  PAD(x, P_n, 3, 3, N);                        // KPAD=32 (8 groups of 4)
  STAT0();
  launch_gemm<32, MODE_H, 64>(stream, P_n, nW1, nb1, 3, N, H_node, stats, nullptr, nullptr);
  FIN(N);
  BN(H_node, P_n, N);
  launch_gemm<64, MODE_Y, 64>(stream, P_n, nW2, nb2, 64, N, node_emb, nullptr, nullptr, nullptr);

  // --- edge embedding MLPs (independent BN stats per edge type) ------------
  {
    const float* eas2[2] = { ea_c, ea_n };
    float* eembs2[2] = { eemb_c, eemb_n };
    for (int i = 0; i < 2; ++i) {
      PAD(eas2[i], P_e64, 4, 3, E);            // KPAD=32
      STAT0();
      launch_gemm<32, MODE_H, 64>(stream, P_e64, eW1, eb1, 4, E, H_edge, stats, nullptr, nullptr);
      FIN(E);
      BN(H_edge, P_e64, E);
      launch_gemm<64, MODE_Y, 64>(stream, P_e64, eW2, eb2, 64, E, eembs2[i], nullptr, nullptr, nullptr);
    }
  }

  float*       eemb[2] = { eemb_c, eemb_n };
  const float* eas[2]  = { ea_c, ea_n };
  const int*   eis[2]  = { ei_c, ei_n };
  float*       sums[2] = { sum_c, sum_n };

  // --- message passing steps -----------------------------------------------
  for (int t = 0; t < S; ++t) {
    Z(sum_c, (size_t)N * 64);
    Z(sum_n, (size_t)N * 64);
    for (int i = 0; i < 2; ++i) {
      int ti = t * 2 + i;
      // message MLP (K=136)
      {
        size_t total4 = (size_t)E * 40;
        pack_msg_kernel<<<(unsigned)((total4 + 255) / 256), 256, 0, stream>>>(
            node_emb, eis[i], eas[i], eemb[i], P_e160, total4);
      }
      STAT0();
      launch_gemm<160, MODE_H, 64>(stream, P_e160, msgW1 + (size_t)ti * 136 * 64,
                                   msgb1 + ti * 64, 136, E, H_edge, stats, nullptr, nullptr);
      FIN(E);
      BN(H_edge, P_e64, E);
      launch_gemm<64, MODE_SCAT, 64>(stream, P_e64, msgW2 + (size_t)ti * 64 * 64,
                                     msgb2 + ti * 64, 64, E, msg, nullptr,
                                     eis[i] + E, sums[i]);
      // edge updater MLP: msg -> new eemb (overwrite; old eemb consumed)
      CAST(msg, P_e64, (size_t)E * 64);
      STAT0();
      launch_gemm<64, MODE_H, 64>(stream, P_e64, edgW1 + (size_t)ti * 64 * 64,
                                  edgb1 + ti * 64, 64, E, H_edge, stats, nullptr, nullptr);
      FIN(E);
      BN(H_edge, P_e64, E);
      launch_gemm<64, MODE_Y, 64>(stream, P_e64, edgW2 + (size_t)ti * 64 * 64,
                                  edgb2 + ti * 64, 64, E, eemb[i], nullptr, nullptr, nullptr);
    }
    // aggr = 0.5 * (mean_c + mean_n)
    aggr_kernel<<<(N * 64 + 255) / 256, 256, 0, stream>>>(sum_c, cnt_c, sum_n, cnt_n,
                                                          aggr, N);
    // node updater: [node_emb | aggr] (K=128) -> node_emb
    {
      size_t total4 = (size_t)N * 32;
      pack_upd_kernel<<<(unsigned)((total4 + 255) / 256), 256, 0, stream>>>(node_emb, aggr,
                                                                            P_n, total4);
    }
    STAT0();
    launch_gemm<128, MODE_H, 64>(stream, P_n, updW1 + (size_t)t * 128 * 64,
                                 updb1 + t * 64, 128, N, H_node, stats, nullptr, nullptr);
    FIN(N);
    BN(H_node, P_n, N);
    launch_gemm<64, MODE_Y, 64>(stream, P_n, updW2 + (size_t)t * 64 * 64,
                                updb2 + t * 64, 64, N, node_emb, nullptr, nullptr, nullptr);
    // decoder: node_emb -> out += [N,3]
    CAST(node_emb, P_n, (size_t)N * 64);
    STAT0();
    launch_gemm<64, MODE_H, 64>(stream, P_n, decW1 + (size_t)t * 64 * 64,
                                decb1 + t * 64, 64, N, H_node, stats, nullptr, nullptr);
    FIN(N);
    BN(H_node, P_n, N);
    launch_gemm<64, MODE_DEC, 3>(stream, P_n, decW2 + (size_t)t * 64 * 3,
                                 decb2 + t * 3, 64, N, out, nullptr, nullptr, nullptr);
  }
}
